// resnext_block_a_6339371729593
// MI455X (gfx1250) — matrix operand order repaired and checked by reading, not compiled, not run on hardware
//
#include <hip/hip_runtime.h>
#include <hip/hip_bf16.h>

#define EPS 1e-3f

typedef _Float16 v16h __attribute__((ext_vector_type(16)));
typedef _Float16 v4h  __attribute__((ext_vector_type(4)));
typedef float    v8f  __attribute__((ext_vector_type(8)));
typedef float    v2f  __attribute__((ext_vector_type(2)));

// Problem geometry
constexpr int IMG_N = 8;
constexpr int IMG_H = 56;
constexpr int IMG_W = 56;
constexpr int CCH   = 256;   // block channels
constexpr int PD    = 128;   // P*D bottleneck width (32 paths * 4)

// Tile geometry: 4 (h) x 16 (w) pixels per workgroup, 3x3 halo -> 6x18
constexpr int TH    = 4;
constexpr int TW    = 16;
constexpr int HROWS = 6;
constexpr int HCOLS = 18;
constexpr int HPX   = HROWS * HCOLS;   // 108 halo pixels
constexpr int MGRP  = 7;               // ceil(108/16) 16-row WMMA M-groups
constexpr int S1    = 132;             // h1 LDS row stride (halves), padded vs 128
constexpr int S2    = 132;             // h2 LDS row stride (floats), padded vs 128

// ---------------------------------------------------------------------------
// Prep kernel: fold BN into weights/shifts, pack WMMA fragments.
//
// w1pack layout (f16 B-fragments for V_WMMA_F32_16X16X32_F16, N=p*4+d, K=c):
//   frag (kc in 0..7, nt in 0..7): lane L, half i:
//     lanes 0-15 : column N = nt*16+L,    K = kc*32 + i        (i = 0..15)
//     lanes 16-31: column N = nt*16+L-16, K = kc*32 + 16 + i
//
// w3pack layout (f32 B-fragments for V_WMMA_F32_16X16X4_F32, per path p,
// per 16-channel slice sl): lane L, float j (j=0,1):
//     lanes 0-15 : N = sl*16+L,    K = j       (VGPR0=K0, VGPR1=K1)
//     lanes 16-31: N = sl*16+L-16, K = 2 + j   (VGPR0=K2, VGPR1=K3)
// BN scale is folded into W1/W2/W3; bias+mean+beta folded into shifts.
// ---------------------------------------------------------------------------
__global__ void resnext_prep(
    const float* __restrict__ W1, const float* __restrict__ B1,
    const float* __restrict__ W2, const float* __restrict__ B2,
    const float* __restrict__ W3, const float* __restrict__ B3,
    const float* __restrict__ G1, const float* __restrict__ Be1,
    const float* __restrict__ M1, const float* __restrict__ V1,
    const float* __restrict__ G2, const float* __restrict__ Be2,
    const float* __restrict__ M2, const float* __restrict__ V2,
    const float* __restrict__ G3, const float* __restrict__ Be3,
    const float* __restrict__ M3, const float* __restrict__ V3,
    _Float16* __restrict__ w1pack, float* __restrict__ shift1,
    float* __restrict__ w2f,      float* __restrict__ shift2,
    float* __restrict__ w3pack,   float* __restrict__ shift3)
{
  const int tid = threadIdx.x;

  // shift1 / shift2 over (P,D)=128
  for (int i = tid; i < PD; i += 256) {
    float s1 = G1[i] * rsqrtf(V1[i] + EPS);
    shift1[i] = Be1[i] + s1 * (B1[i] - M1[i]);
    float s2 = G2[i] * rsqrtf(V2[i] + EPS);
    shift2[i] = Be2[i] + s2 * (B2[i] - M2[i]);
  }

  // w1pack: idx = ((kc*8+nt)*32+L)*16 + i
  for (int idx = tid; idx < 8 * 8 * 32 * 16; idx += 256) {
    int i  = idx & 15;
    int L  = (idx >> 4) & 31;
    int nt = (idx >> 9) & 7;
    int kc = idx >> 12;
    int k    = kc * 32 + ((L < 16) ? i : 16 + i);
    int ncol = nt * 16 + (L & 15);
    int p = ncol >> 2, d = ncol & 3;
    float s1 = G1[ncol] * rsqrtf(V1[ncol] + EPS);
    w1pack[idx] = (_Float16)(W1[(p * CCH + k) * 4 + d] * s1);
  }

  // w2f: (P,3,3,4,4) with scale2 folded into output channel d
  for (int idx = tid; idx < 32 * 144; idx += 256) {
    int d = idx & 3;
    int p = idx / 144;
    float s2 = G2[p * 4 + d] * rsqrtf(V2[p * 4 + d] + EPS);
    w2f[idx] = W2[idx] * s2;
  }

  // shift3 over (P,C)=8192
  for (int idx = tid; idx < 32 * CCH; idx += 256) {
    float s3 = G3[idx] * rsqrtf(V3[idx] + EPS);
    shift3[idx] = Be3[idx] + s3 * (B3[idx] - M3[idx]);
  }

  // w3pack: idx = ((p*16+sl)*32+L)*2 + j
  for (int idx = tid; idx < 32 * 16 * 32 * 2; idx += 256) {
    int j  = idx & 1;
    int L  = (idx >> 1) & 31;
    int sl = (idx >> 6) & 15;
    int p  = idx >> 10;
    int k = j + ((L < 16) ? 0 : 2);
    int c = sl * 16 + (L & 15);
    float s3 = G3[p * CCH + c] * rsqrtf(V3[p * CCH + c] + EPS);
    w3pack[idx] = W3[(p * 4 + k) * CCH + c] * s3;
  }
}

// ---------------------------------------------------------------------------
// Fused ResNeXt block kernel. 256 threads (8 wave32), one 4x16 pixel tile.
// ---------------------------------------------------------------------------
__global__ __launch_bounds__(256) void resnext_main(
    const float* __restrict__ x,
    const _Float16* __restrict__ w1pack,
    const float* __restrict__ shift1,
    const float* __restrict__ w2f,
    const float* __restrict__ shift2,
    const float* __restrict__ w3pack,
    const float* __restrict__ shift3,
    float* __restrict__ out)
{
  __shared__ __align__(16) _Float16 h1s[(MGRP * 16) * S1]; // 112*132 f16 = 29.6KB
  __shared__ __align__(16) float    h2s[(TH * TW) * S2];   //  64*132 f32 = 33.8KB

  const int tid  = threadIdx.x;
  const int lane = tid & 31;
  const int wave = tid >> 5;

  const int wt  = blockIdx.x & 3;                 // 4 tiles across W (last masked)
  const int ht  = (blockIdx.x >> 2) % 14;         // 14 tiles across H
  const int img = blockIdx.x / 56;                // 8 images
  const int h0  = ht * TH;
  const int w0  = wt * TW;

  // ========== Stage 1: h1 = relu(x . (W1*s1) + shift1), f16 WMMA ==========
  // M = 112 (halo pixels, 7 groups of 16), N = 128, K = 256.
  // Each wave owns one N-slice (nt = wave); B fragments hoisted in registers.
  {
    const int nt = wave;
    v16h Bf[8];
#pragma unroll
    for (int kc = 0; kc < 8; ++kc)
      Bf[kc] = *(const v16h*)(w1pack + (size_t)((kc * 8 + nt) * 32 + lane) * 16);

    const float sh  = shift1[nt * 16 + (lane & 15)];
    const int kofs  = (lane < 16) ? 0 : 16;   // contiguous K order (k = 16h + s) offset

    for (int mg = 0; mg < MGRP; ++mg) {
      // A fragment: row M = mg*16 + (lane&15); clamp OOB pixel reads (zeroed later)
      const int mr  = mg * 16 + (lane & 15);
      const int hr  = h0 - 1 + mr / HCOLS;
      const int wc_ = w0 - 1 + mr % HCOLS;
      const int hcl = hr  < 0 ? 0 : (hr  > IMG_H - 1 ? IMG_H - 1 : hr);
      const int wcl = wc_ < 0 ? 0 : (wc_ > IMG_W - 1 ? IMG_W - 1 : wc_);
      const float* xr = x + (size_t)((img * IMG_H + hcl) * IMG_W + wcl) * CCH;

      v16h A[8];
#pragma unroll
      for (int kc = 0; kc < 8; ++kc) {
        // lane<16: K = kc*32 + {0..15}; lane>=16: K = kc*32 + {16..31}
        const float4* s = (const float4*)(xr + kc * 32 + kofs);
        float4 q0 = s[0], q1 = s[1], q2 = s[2], q3 = s[3];
        v16h a;
        a[0]  = (_Float16)q0.x; a[1]  = (_Float16)q0.y;
        a[2]  = (_Float16)q0.z; a[3]  = (_Float16)q0.w;
        a[4]  = (_Float16)q1.x; a[5]  = (_Float16)q1.y;
        a[6]  = (_Float16)q1.z; a[7]  = (_Float16)q1.w;
        a[8]  = (_Float16)q2.x; a[9]  = (_Float16)q2.y;
        a[10] = (_Float16)q2.z; a[11] = (_Float16)q2.w;
        a[12] = (_Float16)q3.x; a[13] = (_Float16)q3.y;
        a[14] = (_Float16)q3.z; a[15] = (_Float16)q3.w;
        A[kc] = a;
      }

      v8f acc = {0.f, 0.f, 0.f, 0.f, 0.f, 0.f, 0.f, 0.f};
#pragma unroll
      for (int kc = 0; kc < 8; ++kc)
        acc = __builtin_amdgcn_wmma_f32_16x16x32_f16(
            false, A[kc], false, Bf[kc], (short)0, acc, false, false);

      // Epilogue: D rows: VGPR j -> M = j (lanes 0-15) / j+8 (lanes 16-31).
      // SAME padding: out-of-image halo pixels must yield h1 == 0.
      const int rbase = mg * 16 + ((lane < 16) ? 0 : 8);
#pragma unroll
      for (int j = 0; j < 8; ++j) {
        const int m2 = rbase + j;
        if (m2 < HPX) {
          const int h2r = h0 - 1 + m2 / HCOLS;
          const int w2c = w0 - 1 + m2 % HCOLS;
          const bool in = (h2r >= 0) & (h2r < IMG_H) & (w2c >= 0) & (w2c < IMG_W);
          const float hv = in ? fmaxf(acc[j] + sh, 0.f) : 0.f;
          h1s[m2 * S1 + nt * 16 + (lane & 15)] = (_Float16)hv;
        }
      }
    }
  }
  __syncthreads();

  // ========== Stage 2: grouped 3x3 conv 4->4 per path, f32 VALU ==========
  // 256 threads = 64 pixels x 4 path-quarters (8 paths each, wave-uniform p).
  {
    const int px   = tid & 63;
    const int p0   = (tid >> 6) * 8;
    const int prow = px >> 4;
    const int pcol = px & 15;
    for (int p = p0; p < p0 + 8; ++p) {
      float a0 = 0.f, a1 = 0.f, a2 = 0.f, a3 = 0.f;
      const float* wp = w2f + p * 144;
#pragma unroll
      for (int dy = 0; dy < 3; ++dy) {
#pragma unroll
        for (int dx = 0; dx < 3; ++dx) {
          const v4h hv = *(const v4h*)(h1s + ((prow + dy) * HCOLS + (pcol + dx)) * S1 + p * 4);
          const float* wk = wp + (dy * 3 + dx) * 16;
#pragma unroll
          for (int e = 0; e < 4; ++e) {
            const float ev = (float)hv[e];
            a0 = fmaf(ev, wk[e * 4 + 0], a0);
            a1 = fmaf(ev, wk[e * 4 + 1], a1);
            a2 = fmaf(ev, wk[e * 4 + 2], a2);
            a3 = fmaf(ev, wk[e * 4 + 3], a3);
          }
        }
      }
      float4 r;
      r.x = fmaxf(a0 + shift2[p * 4 + 0], 0.f);
      r.y = fmaxf(a1 + shift2[p * 4 + 1], 0.f);
      r.z = fmaxf(a2 + shift2[p * 4 + 2], 0.f);
      r.w = fmaxf(a3 + shift2[p * 4 + 3], 0.f);
      *(float4*)(h2s + px * S2 + p * 4) = r;   // stored in f32 A-fragment order
    }
  }
  __syncthreads();

  // ========== Stage 3: per-path 1x1 conv 4->256 via V_WMMA_F32_16X16X4_F32 ==
  // K=4 matches the bottleneck width exactly. Per-path relu forces per-path
  // materialization: acc += relu(A x (W3*s3) + shift3). Then residual + relu.
  {
    for (int u = wave; u < (TH * TW / 16) * 16; u += 8) {   // 4 groups x 16 slices
      const int g  = u >> 4;
      const int sl = u & 15;
      v8f acc = {0.f, 0.f, 0.f, 0.f, 0.f, 0.f, 0.f, 0.f};
      const int arow  = g * 16 + (lane & 15);       // A row (pixel in tile)
      const int kofs3 = (lane < 16) ? 0 : 2;        // ISA 32-bit A layout K offset
#pragma unroll 4
      for (int p = 0; p < 32; ++p) {
        const v2f af = *(const v2f*)(h2s + arow * S2 + p * 4 + kofs3);
        const v2f bf = *(const v2f*)(w3pack + (size_t)((p * 16 + sl) * 32 + lane) * 2);
        const float sh = shift3[p * CCH + sl * 16 + (lane & 15)];
        const v8f cf = {sh, sh, sh, sh, sh, sh, sh, sh};  // shift via C operand
        const v8f dd = __builtin_amdgcn_wmma_f32_16x16x4_f32(
            false, af, false, bf, (short)0, cf, false, false);
#pragma unroll
        for (int j = 0; j < 8; ++j) acc[j] += fmaxf(dd[j], 0.f);
      }
      // residual add + final relu + store
      const int c    = sl * 16 + (lane & 15);
      const int rofs = (lane < 16) ? 0 : 8;
#pragma unroll
      for (int j = 0; j < 8; ++j) {
        const int px  = g * 16 + rofs + j;
        const int hh  = h0 + (px >> 4);
        const int wwp = w0 + (px & 15);
        if (wwp < IMG_W) {
          const size_t o = (size_t)((img * IMG_H + hh) * IMG_W + wwp) * CCH + c;
          out[o] = fmaxf(acc[j] + x[o], 0.f);
        }
      }
    }
  }
}

// ---------------------------------------------------------------------------
extern "C" void kernel_launch(void* const* d_in, const int* in_sizes, int n_in,
                              void* d_out, int out_size, void* d_ws, size_t ws_size,
                              hipStream_t stream) {
  (void)in_sizes; (void)n_in; (void)out_size; (void)ws_size;
  const float* x   = (const float*)d_in[0];
  const float* W1  = (const float*)d_in[1];
  const float* B1  = (const float*)d_in[2];
  const float* W2  = (const float*)d_in[3];
  const float* B2  = (const float*)d_in[4];
  const float* W3  = (const float*)d_in[5];
  const float* B3  = (const float*)d_in[6];
  const float* G1  = (const float*)d_in[7];
  const float* Be1 = (const float*)d_in[8];
  const float* M1  = (const float*)d_in[9];
  const float* V1  = (const float*)d_in[10];
  const float* G2  = (const float*)d_in[11];
  const float* Be2 = (const float*)d_in[12];
  const float* M2  = (const float*)d_in[13];
  const float* V2  = (const float*)d_in[14];
  const float* G3  = (const float*)d_in[15];
  const float* Be3 = (const float*)d_in[16];
  const float* M3  = (const float*)d_in[17];
  const float* V3  = (const float*)d_in[18];

  // Workspace layout (bytes):
  //   [0,      65536)  w1pack  : 32768 f16
  //   [65536,  66048)  shift1  : 128 f32
  //   [66048,  84480)  w2f     : 4608 f32
  //   [84480,  84992)  shift2  : 128 f32
  //   [84992, 216064)  w3pack  : 32768 f32
  //   [216064,248832)  shift3  : 8192 f32
  char* ws = (char*)d_ws;
  _Float16* w1pack = (_Float16*)ws;
  float* shift1    = (float*)(ws + 65536);
  float* w2f       = (float*)(ws + 66048);
  float* shift2    = (float*)(ws + 84480);
  float* w3pack    = (float*)(ws + 84992);
  float* shift3    = (float*)(ws + 216064);

  resnext_prep<<<dim3(1), dim3(256), 0, stream>>>(
      W1, B1, W2, B2, W3, B3,
      G1, Be1, M1, V1, G2, Be2, M2, V2, G3, Be3, M3, V3,
      w1pack, shift1, w2f, shift2, w3pack, shift3);

  // 8 images * 14 h-tiles * 4 w-tiles = 448 workgroups
  resnext_main<<<dim3(448), dim3(256), 0, stream>>>(
      x, w1pack, shift1, w2f, shift2, w3pack, shift3, (float*)d_out);
}
